// MYLayer_28269474742853
// MI455X (gfx1250) — matrix operand order repaired and checked by reading, not compiled, not run on hardware
//
#include <hip/hip_runtime.h>
#include <hip/hip_bf16.h>

#define DI __device__ __forceinline__

typedef __attribute__((ext_vector_type(16))) __bf16 v16bf;
typedef __attribute__((ext_vector_type(8)))  float  v8f;
typedef __attribute__((ext_vector_type(4)))  int    v4i;

#if defined(__has_builtin)
#if __has_builtin(__builtin_amdgcn_global_load_async_to_lds_b128) && \
    __has_builtin(__builtin_amdgcn_s_wait_asynccnt)
#define USE_ASYNC_LDS 1
#endif
#endif
#ifndef USE_ASYNC_LDS
#define USE_ASYNC_LDS 0
#endif

namespace {

constexpr int Bn = 8, Nn = 1024, Hn = 1024, NHn = 16;
constexpr int Mrows = Bn * Nn;   // 8192 rows for every big GEMM

union FragU { uint4 u[2]; v16bf v; };

DI v8f wmma_bf16(v16bf a, v16bf b, v8f c) {
  // D = A(16x32 bf16) * B(32x16 bf16) + C(16x16 f32)
  return __builtin_amdgcn_wmma_f32_16x16x32_bf16(false, a, false, b,
                                                 (short)0, c, false, false);
}

// 16-byte global -> LDS copy. CDNA5 path: async DMA tracked by ASYNCcnt.
// Builtin expects (v4i addrspace(1)* src, v4i addrspace(3)* dst, imm, imm).
DI void cp16(__bf16* l, const __bf16* g) {
#if USE_ASYNC_LDS
  typedef __attribute__((address_space(1))) v4i* gptr_t;
  typedef __attribute__((address_space(3))) v4i* lptr_t;
  __builtin_amdgcn_global_load_async_to_lds_b128(
      (gptr_t)(uintptr_t)g, (lptr_t)(unsigned)(uintptr_t)l, 0, 0);
#else
  *(uint4*)l = *(const uint4*)g;
#endif
}

template <int Nw>
DI void async_wait() {
#if USE_ASYNC_LDS
  __builtin_amdgcn_s_wait_asynccnt(Nw);
#endif
}

// A fragment (16x32, bf16, row-major source with stride ld):
// lane<16: row=lane, elems 0..7 -> K=0..7, elems 8..15 -> K=16..23
// lane>=16: same row, K offset +8 (per ISA layout table)
DI v16bf load_a_frag(const __bf16* base, int row, int ld, int k0, int lane) {
  int r    = row + (lane & 15);
  int koff = (lane >> 4) ? 8 : 0;
  const __bf16* p = base + (size_t)r * ld + k0 + koff;
  FragU f;
  f.u[0] = *(const uint4*)(p);
  f.u[1] = *(const uint4*)(p + 16);
  return f.v;
}

// B fragment (32x16, bf16) from a TRANSPOSED tile Bt[n][k] with stride ld:
// lane<16: col=lane, elems 0..7 -> K=0..7, elems 8..15 -> K=16..23 ; lane>=16: K offset +8
DI v16bf load_b_frag(const __bf16* base, int col, int ld, int k0, int lane) {
  int n    = col + (lane & 15);
  int koff = (lane >> 4) ? 8 : 0;
  const __bf16* p = base + (size_t)n * ld + k0 + koff;
  FragU f;
  f.u[0] = *(const uint4*)(p);
  f.u[1] = *(const uint4*)(p + 16);
  return f.v;
}

DI float gelu_f(float x) { return 0.5f * x * (1.0f + erff(x * 0.70710678118654752f)); }

// ---------------------------------------------------------------------------
// fp32 -> bf16 conversion, optional output row stride / column offset
// (packs prev_query/prev_key straight into the concat buffers).
// ---------------------------------------------------------------------------
__global__ void convert_k(const float* __restrict__ src, int sld,
                          __bf16* __restrict__ dst, int dld, int doff,
                          long long rows, int cols) {
  long long total = rows * (long long)cols;
  for (long long i = (long long)blockIdx.x * blockDim.x + threadIdx.x; i < total;
       i += (long long)gridDim.x * blockDim.x) {
    long long r = i / cols;
    int c = (int)(i - r * cols);
    dst[r * dld + doff + c] = (__bf16)src[r * sld + c];
  }
}

// fp32 [rows][cols] -> bf16 TRANSPOSED [cols][rows] (weights, one-time)
__global__ void convertT_k(const float* __restrict__ src, int rows, int cols,
                           __bf16* __restrict__ dst) {
  long long total = (long long)rows * cols;
  for (long long i = (long long)blockIdx.x * blockDim.x + threadIdx.x; i < total;
       i += (long long)gridDim.x * blockDim.x) {
    long long r = i / cols;
    int c = (int)(i - r * cols);
    dst[(size_t)c * rows + r] = (__bf16)src[i];
  }
}

// ---------------------------------------------------------------------------
// Tiled bf16 WMMA GEMM: C[8192 x Nc] = A[8192 x K] * Wt^T (+ epilogue).
// Wt is PRE-TRANSPOSED [Nc][K] so both LDS tiles are contiguous copies.
// Block tile 128x128, 8 waves x (32x64), K-step 32, double-buffered LDS with
// async global->LDS copies overlapping WMMA of the previous tile.
// ---------------------------------------------------------------------------
template <bool MASKROW, bool GELUA, bool RESID, bool TRANSH>
__global__ __launch_bounds__(256)
void gemm_bf16_k(const __bf16* __restrict__ A, int lda,
                 const __bf16* __restrict__ Wt, int Nc, int K,
                 const float* __restrict__ bias,
                 const unsigned char* __restrict__ rowmask,
                 const float* __restrict__ resid,
                 float* __restrict__ outf, int ldf,
                 __bf16* __restrict__ outh, int ldh) {
  __shared__ __align__(16) __bf16 Asl[2][128 * 40];  // [row][k] padded
  __shared__ __align__(16) __bf16 Btl[2][128 * 40];  // [n][k]   padded

  const int tid  = threadIdx.x;
  const int lane = tid & 31;
  const int w    = tid >> 5;
  const int wm   = w & 3, wn = w >> 2;
  const int m0   = blockIdx.y * 128;
  const int n0   = blockIdx.x * 128;

  auto issue_tile = [&](int kt, int buf) {
#pragma unroll
    for (int i = 0; i < 2; ++i) {
      int j = tid + i * 256;                     // 0..511
      int row = j >> 2, seg = (j & 3) * 8;       // 4 x 16B chunks per 32-half row
      cp16(&Asl[buf][row * 40 + seg], A  + (size_t)(m0 + row) * lda + kt * 32 + seg);
      cp16(&Btl[buf][row * 40 + seg], Wt + (size_t)(n0 + row) * K   + kt * 32 + seg);
    }
  };

  v8f c[2][4];
#pragma unroll
  for (int i = 0; i < 2; ++i)
#pragma unroll
    for (int j = 0; j < 4; ++j)
#pragma unroll
      for (int r = 0; r < 8; ++r) c[i][j][r] = 0.0f;

  const int ktiles = K >> 5;
  issue_tile(0, 0);
  for (int kt = 0; kt < ktiles; ++kt) {
    const int cur = kt & 1;
    if (kt + 1 < ktiles) {
      issue_tile(kt + 1, cur ^ 1);  // prefetch next tile into other buffer
      async_wait<4>();              // tile kt's 4 copies done; next 4 in flight
    } else {
      async_wait<0>();
    }
    __syncthreads();

    const __bf16* As = Asl[cur];
    const __bf16* Bs = Btl[cur];
    v16bf a0 = load_a_frag(As, wm * 32,      40, 0, lane);
    v16bf a1 = load_a_frag(As, wm * 32 + 16, 40, 0, lane);
    v16bf b0 = load_b_frag(Bs, wn * 64 +  0, 40, 0, lane);
    v16bf b1 = load_b_frag(Bs, wn * 64 + 16, 40, 0, lane);
    v16bf b2 = load_b_frag(Bs, wn * 64 + 32, 40, 0, lane);
    v16bf b3 = load_b_frag(Bs, wn * 64 + 48, 40, 0, lane);
    c[0][0] = wmma_bf16(a0, b0, c[0][0]);
    c[0][1] = wmma_bf16(a0, b1, c[0][1]);
    c[0][2] = wmma_bf16(a0, b2, c[0][2]);
    c[0][3] = wmma_bf16(a0, b3, c[0][3]);
    c[1][0] = wmma_bf16(a1, b0, c[1][0]);
    c[1][1] = wmma_bf16(a1, b1, c[1][1]);
    c[1][2] = wmma_bf16(a1, b2, c[1][2]);
    c[1][3] = wmma_bf16(a1, b3, c[1][3]);
    __syncthreads();  // buffer 'cur' free for reuse at tile kt+2
  }

  const int hlf = lane >> 4, ln = lane & 15;
#pragma unroll
  for (int fm = 0; fm < 2; ++fm)
#pragma unroll
    for (int fn = 0; fn < 4; ++fn)
#pragma unroll
      for (int r = 0; r < 8; ++r) {
        int row = m0 + wm * 32 + fm * 16 + r + hlf * 8;  // C layout: M = r (+8 hi)
        int col = n0 + wn * 64 + fn * 16 + ln;           //           N = lane&15
        float v = c[fm][fn][r] + bias[col];
        if (MASKROW) v *= (float)rowmask[row];
        if (RESID)   v += resid[(size_t)row * Nc + col];
        if (GELUA)   v = gelu_f(v);
        if (outf) outf[(size_t)row * ldf + col] = v;
        if (outh) {
          if (TRANSH) {  // scatter into [B, H, N] (transposed V for attention)
            int bb = row >> 10, nloc = row & (Nn - 1);
            outh[((size_t)bb * Hn + col) * Nn + nloc] = (__bf16)v;
          } else {
            outh[(size_t)row * ldh + col] = (__bf16)v;
          }
        }
      }
}

// ---------------------------------------------------------------------------
// Flash attention with prior bias + column mask.
// Grid: (N/64, NH, B). Block: 128 threads = 4 waves; each wave owns 16 rows.
// K tiles copied straight; V comes pre-transposed [B,H,N] so its tile is a
// straight copy too. Double-buffered with async copies.
// ---------------------------------------------------------------------------
__global__ __launch_bounds__(128)
void flash_attn_k(const __bf16* __restrict__ Q, const __bf16* __restrict__ Kb,
                  const __bf16* __restrict__ Vtg,  // [B, H, N]
                  const float* __restrict__ prior,
                  const unsigned char* __restrict__ mask,
                  __bf16* __restrict__ ctx) {
  __shared__ __align__(16) __bf16 KtL[2][64 * 72];  // [key][d]
  __shared__ __align__(16) __bf16 VtL[2][64 * 72];  // [d][key]
  __shared__ __align__(16) __bf16 Pl[4 * 16 * 72];  // per-wave P staging

  const int mt = blockIdx.x, h = blockIdx.y, b = blockIdx.z;
  const int tid = threadIdx.x, lane = tid & 31, w = tid >> 5;
  const int hlf = lane >> 4, ln = lane & 15;
  __bf16* Pw = Pl + w * (16 * 72);

  auto issue_kv = [&](int nt, int buf) {
    const __bf16* kg = Kb  + ((size_t)(b * Nn + nt * 64)) * Hn + h * 64;
    const __bf16* vg = Vtg + ((size_t)(b * Hn + h * 64)) * Nn + nt * 64;
#pragma unroll
    for (int i = 0; i < 4; ++i) {
      int j = tid + i * 128;                 // 0..511
      int row = j >> 3, c8 = (j & 7) * 8;    // 8 x 16B chunks per 64-half row
      cp16(&KtL[buf][row * 72 + c8], kg + (size_t)row * Hn + c8);
      cp16(&VtL[buf][row * 72 + c8], vg + (size_t)row * Nn + c8);
    }
  };

  // Q fragments for this wave's 16 rows (d = 0..63 -> two A frags)
  v16bf qa[2];
  {
    const __bf16* qp =
        Q + ((size_t)(b * Nn + mt * 64 + w * 16 + ln)) * Hn + h * 64;
    int koff = hlf ? 8 : 0;
#pragma unroll
    for (int ks = 0; ks < 2; ++ks) {
      FragU f;
      f.u[0] = *(const uint4*)(qp + ks * 32 + koff);
      f.u[1] = *(const uint4*)(qp + ks * 32 + 16 + koff);
      qa[ks] = f.v;
    }
  }

  float mrow[8], lrow[8];
  v8f o[4];
#pragma unroll
  for (int r = 0; r < 8; ++r) { mrow[r] = -3.0e38f; lrow[r] = 0.0f; }
#pragma unroll
  for (int d = 0; d < 4; ++d)
#pragma unroll
    for (int r = 0; r < 8; ++r) o[d][r] = 0.0f;

  constexpr int NT = Nn / 64;
  issue_kv(0, 0);
  for (int nt = 0; nt < NT; ++nt) {
    const int cur = nt & 1;
    if (nt + 1 < NT) {
      issue_kv(nt + 1, cur ^ 1);
      async_wait<8>();
    } else {
      async_wait<0>();
    }
    __syncthreads();

    // S = Q * K^T  (B-frag element e is K[key][d], contiguous in d)
    v8f s[4];
#pragma unroll
    for (int fn = 0; fn < 4; ++fn) {
#pragma unroll
      for (int r = 0; r < 8; ++r) s[fn][r] = 0.0f;
#pragma unroll
      for (int ks = 0; ks < 2; ++ks) {
        v16bf kb = load_b_frag(KtL[cur], fn * 16, 72, ks * 32, lane);
        s[fn] = wmma_bf16(qa[ks], kb, s[fn]);
      }
    }

    // scale + prior + column mask, online softmax
    float mnew[8], sv[4][8];
#pragma unroll
    for (int r = 0; r < 8; ++r) mnew[r] = mrow[r];
#pragma unroll
    for (int fn = 0; fn < 4; ++fn) {
      int col = nt * 64 + fn * 16 + ln;
      float cmask = mask[(size_t)b * Nn + col] ? 0.0f : -3.0e38f;
#pragma unroll
      for (int r = 0; r < 8; ++r) {
        int row = mt * 64 + w * 16 + r + hlf * 8;
        float v = s[fn][r] * 0.125f +
                  prior[((size_t)(b * Nn + row)) * Nn + col] + cmask;
        sv[fn][r] = v;
        mnew[r] = fmaxf(mnew[r], v);
      }
    }
#pragma unroll
    for (int d = 1; d < 16; d <<= 1)
#pragma unroll
      for (int r = 0; r < 8; ++r)
        mnew[r] = fmaxf(mnew[r], __shfl_xor(mnew[r], d, 32));

    float alpha[8], rsum[8];
#pragma unroll
    for (int r = 0; r < 8; ++r) {
      alpha[r] = __expf(mrow[r] - mnew[r]);
      mrow[r] = mnew[r];
      rsum[r] = 0.0f;
    }
#pragma unroll
    for (int fn = 0; fn < 4; ++fn)
#pragma unroll
      for (int r = 0; r < 8; ++r) {
        float pv = __expf(sv[fn][r] - mnew[r]);
        rsum[r] += pv;
        Pw[(r + hlf * 8) * 72 + fn * 16 + ln] = (__bf16)pv;  // C->LDS
      }
#pragma unroll
    for (int d = 1; d < 16; d <<= 1)
#pragma unroll
      for (int r = 0; r < 8; ++r) rsum[r] += __shfl_xor(rsum[r], d, 32);
#pragma unroll
    for (int r = 0; r < 8; ++r) lrow[r] = lrow[r] * alpha[r] + rsum[r];
#pragma unroll
    for (int ds = 0; ds < 4; ++ds)
#pragma unroll
      for (int r = 0; r < 8; ++r) o[ds][r] *= alpha[r];

    __syncthreads();  // P visible; keeps waves in step

    // O += P * V   (A = P reloaded in A-layout, B from transposed V tile)
    v16bf pa0 = load_a_frag(Pw, 0, 72, 0, lane);
    v16bf pa1 = load_a_frag(Pw, 0, 72, 32, lane);
#pragma unroll
    for (int ds = 0; ds < 4; ++ds) {
      v16bf vb0 = load_b_frag(VtL[cur], ds * 16, 72, 0, lane);
      o[ds] = wmma_bf16(pa0, vb0, o[ds]);
      v16bf vb1 = load_b_frag(VtL[cur], ds * 16, 72, 32, lane);
      o[ds] = wmma_bf16(pa1, vb1, o[ds]);
    }
    __syncthreads();  // buffer 'cur' free for tile nt+2
  }

  float inv[8];
#pragma unroll
  for (int r = 0; r < 8; ++r)
    inv[r] = (lrow[r] > 0.0f) ? (1.0f / lrow[r]) : 0.0f;  // nan_to_num
#pragma unroll
  for (int ds = 0; ds < 4; ++ds)
#pragma unroll
    for (int r = 0; r < 8; ++r) {
      int row = mt * 64 + w * 16 + r + hlf * 8;
      int d = ds * 16 + ln;
      ctx[((size_t)(b * Nn + row)) * Hn + h * 64 + d] =
          (__bf16)(o[ds][r] * inv[r]);
    }
}

// ---------------------------------------------------------------------------
// LayerNorm per row (H=1024), optional post-LN row mask, optional bf16 copy.
// ---------------------------------------------------------------------------
__global__ __launch_bounds__(256)
void layernorm_k(const float* __restrict__ in, const float* __restrict__ gam,
                 const float* __restrict__ bet,
                 const unsigned char* __restrict__ rowmask,
                 float* __restrict__ outf, __bf16* __restrict__ outh) {
  __shared__ float reds[16];
  const int row = blockIdx.x;
  const int tid = threadIdx.x;
  const float* x = in + (size_t)row * Hn;
  float vals[4], s = 0.f, ss = 0.f;
#pragma unroll
  for (int i = 0; i < 4; ++i) {
    vals[i] = x[tid + i * 256];
    s += vals[i];
    ss += vals[i] * vals[i];
  }
#pragma unroll
  for (int d = 1; d < 32; d <<= 1) {
    s += __shfl_xor(s, d, 32);
    ss += __shfl_xor(ss, d, 32);
  }
  if ((tid & 31) == 0) { reds[tid >> 5] = s; reds[8 + (tid >> 5)] = ss; }
  __syncthreads();
  s = 0.f; ss = 0.f;
#pragma unroll
  for (int i = 0; i < 8; ++i) { s += reds[i]; ss += reds[8 + i]; }
  const float mu = s * (1.0f / Hn);
  const float var = ss * (1.0f / Hn) - mu * mu;
  const float rs = rsqrtf(var + 1e-5f);
  const float mf = rowmask ? (float)rowmask[row] : 1.0f;
#pragma unroll
  for (int i = 0; i < 4; ++i) {
    int c = tid + i * 256;
    float y = ((vals[i] - mu) * rs * gam[c] + bet[c]) * mf;
    if (outf) outf[(size_t)row * Hn + c] = y;
    if (outh) outh[(size_t)row * Hn + c] = (__bf16)y;
  }
}

}  // namespace

extern "C" void kernel_launch(void* const* d_in, const int* in_sizes, int n_in,
                              void* d_out, int out_size, void* d_ws,
                              size_t ws_size, hipStream_t stream) {
  (void)in_sizes; (void)n_in; (void)out_size; (void)ws_size;

  const float* x     = (const float*)d_in[0];
  const float* prior = (const float*)d_in[1];
  const unsigned char* mask = (const unsigned char*)d_in[2];
  const float* pq  = (const float*)d_in[3];
  const float* pk  = (const float*)d_in[4];
  const float* Wq  = (const float*)d_in[5];  const float* bq  = (const float*)d_in[6];
  const float* Wk  = (const float*)d_in[7];  const float* bk  = (const float*)d_in[8];
  const float* Wv  = (const float*)d_in[9];  const float* bv  = (const float*)d_in[10];
  const float* Wqf = (const float*)d_in[11]; const float* bqf = (const float*)d_in[12];
  const float* Wkf = (const float*)d_in[13]; const float* bkf = (const float*)d_in[14];
  const float* Wo  = (const float*)d_in[15]; const float* bo  = (const float*)d_in[16];
  const float* g1  = (const float*)d_in[17]; const float* be1 = (const float*)d_in[18];
  const float* W1  = (const float*)d_in[19]; const float* b1  = (const float*)d_in[20];
  const float* W2  = (const float*)d_in[21]; const float* b2  = (const float*)d_in[22];
  const float* g2  = (const float*)d_in[23]; const float* be2 = (const float*)d_in[24];

  float* out_x2 = (float*)d_out;                // [B,N,H]
  float* out_fq = out_x2 + (size_t)Mrows * Hn;  // fused_q
  float* out_fk = out_fq + (size_t)Mrows * Hn;  // fused_k

  char* p = (char*)d_ws;
  auto take = [&](size_t bytes) {
    char* q = p;
    p += (bytes + 255) & ~(size_t)255;
    return q;
  };
  __bf16* xb    = (__bf16*)take((size_t)Mrows * Hn * 2);
  __bf16* WqT   = (__bf16*)take((size_t)Hn * Hn * 2);       // [Nc][K] transposed
  __bf16* WkT   = (__bf16*)take((size_t)Hn * Hn * 2);
  __bf16* WvT   = (__bf16*)take((size_t)Hn * Hn * 2);
  __bf16* WoT   = (__bf16*)take((size_t)Hn * Hn * 2);
  __bf16* WqfT  = (__bf16*)take((size_t)2 * Hn * Hn * 2);
  __bf16* WkfT  = (__bf16*)take((size_t)2 * Hn * Hn * 2);
  __bf16* W1T   = (__bf16*)take((size_t)Hn * 2 * Hn * 2);
  __bf16* W2T   = (__bf16*)take((size_t)2 * Hn * Hn * 2);
  __bf16* cat_q = (__bf16*)take((size_t)Mrows * 2 * Hn * 2);  // reused as h1
  __bf16* cat_k = (__bf16*)take((size_t)Mrows * 2 * Hn * 2);
  __bf16* v_t   = (__bf16*)take((size_t)Mrows * Hn * 2);      // V as [B,H,N]
  __bf16* q_b   = (__bf16*)take((size_t)Mrows * Hn * 2);
  __bf16* k_b   = (__bf16*)take((size_t)Mrows * Hn * 2);
  __bf16* ctx_b = (__bf16*)take((size_t)Mrows * Hn * 2);
  float*  pre1  = (float*)take((size_t)Mrows * Hn * 4);       // reused for pre2
  float*  x1    = (float*)take((size_t)Mrows * Hn * 4);
  __bf16* x1_b  = (__bf16*)take((size_t)Mrows * Hn * 2);
  __bf16* h1    = cat_q;  // alias: cat_q dead after the fused-Q GEMM

  auto conv = [&](const float* s, int sld, __bf16* d, int dld, int doff,
                  long long rows, int cols) {
    long long total = rows * (long long)cols;
    int blocks = (int)((total + 2047) / 2048);
    if (blocks > 4096) blocks = 4096;
    convert_k<<<blocks, 256, 0, stream>>>(s, sld, d, dld, doff, rows, cols);
  };
  auto convT = [&](const float* s, int rows, int cols, __bf16* d) {
    long long total = (long long)rows * cols;
    int blocks = (int)((total + 2047) / 2048);
    if (blocks > 4096) blocks = 4096;
    convertT_k<<<blocks, 256, 0, stream>>>(s, rows, cols, d);
  };

  // --- one-time fp32 -> bf16 conversions (weights also transposed) ---
  conv(x,  Hn, xb,    Hn,     0,  Mrows, Hn);
  conv(pq, Hn, cat_q, 2 * Hn, Hn, Mrows, Hn);
  conv(pk, Hn, cat_k, 2 * Hn, Hn, Mrows, Hn);
  convT(Wq,  Hn,     Hn,     WqT);
  convT(Wk,  Hn,     Hn,     WkT);
  convT(Wv,  Hn,     Hn,     WvT);
  convT(Wo,  Hn,     Hn,     WoT);
  convT(Wqf, 2 * Hn, Hn,     WqfT);
  convT(Wkf, 2 * Hn, Hn,     WkfT);
  convT(W1,  Hn,     2 * Hn, W1T);
  convT(W2,  2 * Hn, Hn,     W2T);

  dim3 blk(256);
  dim3 g1024(Hn / 128, Mrows / 128);      // (8, 64)
  dim3 g2048(2 * Hn / 128, Mrows / 128);  // (16, 64)

  // q/k/v projections (bias + row mask); V scattered into [B,H,N]
  gemm_bf16_k<true, false, false, false><<<g1024, blk, 0, stream>>>(
      xb, Hn, WqT, Hn, Hn, bq, mask, nullptr, nullptr, 0, cat_q, 2 * Hn);
  gemm_bf16_k<true, false, false, false><<<g1024, blk, 0, stream>>>(
      xb, Hn, WkT, Hn, Hn, bk, mask, nullptr, nullptr, 0, cat_k, 2 * Hn);
  gemm_bf16_k<true, false, false, true><<<g1024, blk, 0, stream>>>(
      xb, Hn, WvT, Hn, Hn, bv, mask, nullptr, nullptr, 0, v_t, 0);

  // fused q/k (K = 2048): fp32 straight to d_out + bf16 copies for attention
  gemm_bf16_k<false, false, false, false><<<g1024, blk, 0, stream>>>(
      cat_q, 2 * Hn, WqfT, Hn, 2 * Hn, bqf, nullptr, nullptr, out_fq, Hn, q_b, Hn);
  gemm_bf16_k<false, false, false, false><<<g1024, blk, 0, stream>>>(
      cat_k, 2 * Hn, WkfT, Hn, 2 * Hn, bkf, nullptr, nullptr, out_fk, Hn, k_b, Hn);

  // attention with prior + mask -> ctx (bf16)
  flash_attn_k<<<dim3(Nn / 64, NHn, Bn), dim3(128), 0, stream>>>(
      q_b, k_b, v_t, prior, mask, ctx_b);

  // out proj + residual(x) -> pre1 ; LN1 (+mask) -> x1 (f32 + bf16)
  gemm_bf16_k<false, false, true, false><<<g1024, blk, 0, stream>>>(
      ctx_b, Hn, WoT, Hn, Hn, bo, nullptr, x, pre1, Hn, nullptr, 0);
  layernorm_k<<<Mrows, 256, 0, stream>>>(pre1, g1, be1, mask, x1, x1_b);

  // FFN: gelu(x1@W1+b1) -> h1 ; h1@W2+b2 + x1 -> pre2 ; LN2 -> x2
  gemm_bf16_k<false, true, false, false><<<g2048, blk, 0, stream>>>(
      x1_b, Hn, W1T, 2 * Hn, Hn, b1, nullptr, nullptr, nullptr, 0, h1, 2 * Hn);
  gemm_bf16_k<false, false, true, false><<<g1024, blk, 0, stream>>>(
      h1, 2 * Hn, W2T, Hn, 2 * Hn, b2, nullptr, x1, pre1, Hn, nullptr, 0);
  layernorm_k<<<Mrows, 256, 0, stream>>>(pre1, g2, be2, nullptr, out_x2, nullptr);
}
